// GlobalAttentionTransformer_446676599180
// MI455X (gfx1250) — matrix operand order repaired and checked by reading, not compiled, not run on hardware
//
#include <hip/hip_runtime.h>
#include <math.h>

// ---------------- problem constants ----------------
#define D_MODEL 768
#define NH      12
#define HD      64
#define NREG    4
#define SEQ     1029          // 1 + 4 + 32*32
#define BATCH   8
#define HWSZ    1024          // 32*32
#define MTOT    (BATCH*SEQ)   // 8232 rows

typedef __bf16 bf16;
typedef __attribute__((ext_vector_type(16))) __bf16 v16bf;
typedef __attribute__((ext_vector_type(8)))  __bf16 v8bf;
typedef __attribute__((ext_vector_type(8)))  float  v8f;

// Build a 16-element bf16 WMMA fragment from two 16-byte LDS chunks.
__device__ inline v16bf ldfrag(const bf16* row, int o0, int o1) {
    v8bf lo = *(const v8bf*)(row + o0);
    v8bf hi = *(const v8bf*)(row + o1);
    return __builtin_shufflevector(lo, hi, 0,1,2,3,4,5,6,7,8,9,10,11,12,13,14,15);
}

// =====================================================================
// WMMA GEMM: C[M,N] = A[M,K] * W[N,K]^T + bias[N]  (+ epilogue)
//   EPI 0: store f32
//   EPI 1: store f32 with residual add (Res[M,N])
//   EPI 2: exact GELU, store bf16
// Block tile 128(M) x 128(N), 256 threads = 8 waves.
// Wave w owns rows [16w,16w+16) x all 128 N (8 accumulators).
// Register-staged global loads + double-buffered LDS. An empty asm
// "register fence" pins A + all 8 B fragments in distinct VGPRs so the
// 18 ds_loads issue as one burst with a single s_wait_dscnt, followed
// by 8 back-to-back WMMAs. One barrier per K-step.
// Requires: N % 128 == 0, K % 32 == 0. May over-read <=64B past the last
// K-tile (stays inside workspace; never consumed).
// =====================================================================
template<int EPI>
__global__ __launch_bounds__(256) void gemm_bf16_wmma(
    const bf16* __restrict__ A, const bf16* __restrict__ W,
    const float* __restrict__ bias, const float* __restrict__ Res,
    void* __restrict__ Out, int M, int N, int K)
{
    __shared__ alignas(16) bf16 Al[2][128][40];   // +8 pad: 16B-aligned rows, bank spread
    __shared__ alignas(16) bf16 Bl[2][128][40];

    const int tid  = threadIdx.x;
    const int wave = tid >> 5, lane = tid & 31;
    const int half = lane >> 4, lm = lane & 15;
    const int n0 = blockIdx.x * 128;
    const int m0 = blockIdx.y * 128;

    // Loop-invariant staging assignment: thread owns chunks (r, kc) and (r+64, kc)
    const int r0 = tid >> 2;            // 0..63
    const int r1 = r0 + 64;             // 64..127
    const int kc = (tid & 3) * 8;       // 0/8/16/24
    int gm0 = m0 + r0; if (gm0 >= M) gm0 = M - 1;   // clamp; dead rows dropped on store
    int gm1 = m0 + r1; if (gm1 >= M) gm1 = M - 1;
    const bf16* aPtr0 = A + (size_t)gm0 * K + kc;
    const bf16* aPtr1 = A + (size_t)gm1 * K + kc;
    const bf16* bPtr0 = W + (size_t)(n0 + r0) * K + kc;
    const bf16* bPtr1 = W + (size_t)(n0 + r1) * K + kc;

    // Prologue: stage tile 0 into registers.
    v8bf a0 = *(const v8bf*)(aPtr0);
    v8bf a1 = *(const v8bf*)(aPtr1);
    v8bf b0 = *(const v8bf*)(bPtr0);
    v8bf b1 = *(const v8bf*)(bPtr1);

    v8f acc[8] = {};
    const int nIter = K >> 5;

    for (int it = 0; it < nIter; ++it) {
        const int cur = it & 1;
        // Commit staged registers to LDS (their loads were issued one full
        // WMMA-chain ago, so loadcnt waits are cheap here).
        *(v8bf*)&Al[cur][r0][kc] = a0;
        *(v8bf*)&Al[cur][r1][kc] = a1;
        *(v8bf*)&Bl[cur][r0][kc] = b0;
        *(v8bf*)&Bl[cur][r1][kc] = b1;
        __syncthreads();

        // Advance by a compile-time constant and issue next tile's loads
        // unconditionally (final iteration over-reads <=64B of workspace).
        aPtr0 += 32; aPtr1 += 32; bPtr0 += 32; bPtr1 += 32;
        a0 = *(const v8bf*)aPtr0;
        a1 = *(const v8bf*)aPtr1;
        b0 = *(const v8bf*)bPtr0;
        b1 = *(const v8bf*)bPtr1;

        // A fragment (16x32): lanes 0-15 K {0..7,16..23}, lanes 16-31 {8..15,24..31}
        v16bf a = ldfrag(&Al[cur][wave * 16 + lm][0], half * 8, 16 + half * 8);
        // Preload all 8 B fragments (32x16: lane = column, lanes 0-15 K {0..7,16..23}, lanes 16-31 {8..15,24..31}).
        v16bf bfr[8];
        #pragma unroll
        for (int j = 0; j < 8; ++j)
            bfr[j] = ldfrag(&Bl[cur][j * 16 + lm][0], half * 8, 16 + half * 8);
        // Register fence: force all fragments into distinct live VGPRs so the
        // 18 ds_loads complete under ONE s_wait_dscnt and the 8 WMMAs below
        // issue as an uninterrupted burst (no load/wait interleave).
        asm volatile("" : "+v"(a), "+v"(bfr[0]), "+v"(bfr[1]), "+v"(bfr[2]),
                          "+v"(bfr[3]), "+v"(bfr[4]), "+v"(bfr[5]),
                          "+v"(bfr[6]), "+v"(bfr[7]));
        #pragma unroll
        for (int j = 0; j < 8; ++j)
            acc[j] = __builtin_amdgcn_wmma_f32_16x16x32_bf16(
                false, a, false, bfr[j], (short)0, acc[j], false, false);
        // No trailing barrier needed: a wave reaches the next iteration's
        // barrier only after its fragment ds_loads completed (WMMA issue
        // waits dscnt), and the next store targets the other LDS buffer.
    }

    // D layout: VGPR r -> M = r + 8*half, N = lane&15
    #pragma unroll
    for (int j = 0; j < 8; ++j) {
        const int n = n0 + j * 16 + lm;
        const float bi = bias ? bias[n] : 0.0f;
        #pragma unroll
        for (int r = 0; r < 8; ++r) {
            const int m = m0 + wave * 16 + half * 8 + r;
            if (m >= M) continue;
            float v = acc[j][r] + bi;
            const size_t idx = (size_t)m * N + n;
            if (EPI == 0) {
                ((float*)Out)[idx] = v;
            } else if (EPI == 1) {
                ((float*)Out)[idx] = v + Res[idx];
            } else {
                float g = 0.5f * v * (1.0f + erff(v * 0.70710678118f));
                ((bf16*)Out)[idx] = (bf16)g;
            }
        }
    }
}

// =====================================================================
// Gather (ctx | registers | NCHW->tokens) + LayerNorm1. One block per row.
// Emits fp32 (residual path) and bf16 (GEMM operand).
// =====================================================================
__global__ __launch_bounds__(256) void prep_ln1_kernel(
    const float* __restrict__ x, const float* __restrict__ ctx,
    const float* __restrict__ regs, const float* __restrict__ g,
    const float* __restrict__ bta, float* __restrict__ xw,
    bf16* __restrict__ xwh)
{
    __shared__ float red[256];
    const int row = blockIdx.x;
    const int bb = row / SEQ, s = row % SEQ;
    const int tid = threadIdx.x;

    float v[3];
    #pragma unroll
    for (int i = 0; i < 3; ++i) {
        int c = tid + i * 256;
        float val;
        if (s == 0)      val = ctx[(size_t)bb * D_MODEL + c];
        else if (s < 5)  val = regs[((size_t)bb * NREG + (s - 1)) * D_MODEL + c];
        else             val = x[((size_t)bb * D_MODEL + c) * HWSZ + (s - 5)];
        v[i] = val;
    }
    red[tid] = v[0] + v[1] + v[2]; __syncthreads();
    for (int st = 128; st > 0; st >>= 1) { if (tid < st) red[tid] += red[tid + st]; __syncthreads(); }
    float mu = red[0] * (1.0f / D_MODEL); __syncthreads();

    float d0 = v[0] - mu, d1 = v[1] - mu, d2 = v[2] - mu;
    red[tid] = d0*d0 + d1*d1 + d2*d2; __syncthreads();
    for (int st = 128; st > 0; st >>= 1) { if (tid < st) red[tid] += red[tid + st]; __syncthreads(); }
    float rstd = rsqrtf(red[0] * (1.0f / D_MODEL) + 1e-5f);

    #pragma unroll
    for (int i = 0; i < 3; ++i) {
        int c = tid + i * 256;
        float y = (v[i] - mu) * rstd * g[c] + bta[c];
        xw[(size_t)row * D_MODEL + c]  = y;
        xwh[(size_t)row * D_MODEL + c] = (bf16)y;
    }
}

// ---------------- LayerNorm2 (fp32 in, bf16 out), one block per row ----
__global__ __launch_bounds__(256) void ln2_kernel(
    const float* __restrict__ h1, const float* __restrict__ g,
    const float* __restrict__ bta, bf16* __restrict__ out)
{
    __shared__ float red[256];
    const int row = blockIdx.x;
    const int tid = threadIdx.x;
    float v[3];
    #pragma unroll
    for (int i = 0; i < 3; ++i) v[i] = h1[(size_t)row * D_MODEL + tid + i * 256];
    red[tid] = v[0] + v[1] + v[2]; __syncthreads();
    for (int st = 128; st > 0; st >>= 1) { if (tid < st) red[tid] += red[tid + st]; __syncthreads(); }
    float mu = red[0] * (1.0f / D_MODEL); __syncthreads();
    float d0 = v[0]-mu, d1 = v[1]-mu, d2 = v[2]-mu;
    red[tid] = d0*d0 + d1*d1 + d2*d2; __syncthreads();
    for (int st = 128; st > 0; st >>= 1) { if (tid < st) red[tid] += red[tid + st]; __syncthreads(); }
    float rstd = rsqrtf(red[0] * (1.0f / D_MODEL) + 1e-5f);
    #pragma unroll
    for (int i = 0; i < 3; ++i) {
        int c = tid + i * 256;
        out[(size_t)row * D_MODEL + c] = (bf16)((v[i] - mu) * rstd * g[c] + bta[c]);
    }
}

// =====================================================================
// Attention, feature rows (s>=5): exactly 6 visible keys {0..4, s}.
// One thread per (b, s, h).
// =====================================================================
__global__ __launch_bounds__(256) void attn_feat_kernel(
    const float* __restrict__ qkv, bf16* __restrict__ attn)
{
    int t = blockIdx.x * 256 + threadIdx.x;
    if (t >= BATCH * HWSZ * NH) return;
    int h = t % NH; int idx = t / NH;
    int s = 5 + (idx % HWSZ); int bb = idx / HWSZ;

    const float* q = qkv + ((size_t)(bb * SEQ + s)) * 3 * D_MODEL + h * HD;
    int keys[6] = {0, 1, 2, 3, 4, s};
    float sc[6];
    #pragma unroll
    for (int j = 0; j < 6; ++j) {
        const float* kp = qkv + ((size_t)(bb * SEQ + keys[j])) * 3 * D_MODEL + D_MODEL + h * HD;
        float d = 0.0f;
        for (int c = 0; c < HD; ++c) d += q[c] * kp[c];
        sc[j] = d * 0.125f;             // 1/sqrt(64)
    }
    float mx = sc[0];
    #pragma unroll
    for (int j = 1; j < 6; ++j) mx = fmaxf(mx, sc[j]);
    float se = 0.0f;
    #pragma unroll
    for (int j = 0; j < 6; ++j) { sc[j] = expf(sc[j] - mx); se += sc[j]; }
    float inv = 1.0f / se;

    float acc[HD];
    for (int c = 0; c < HD; ++c) acc[c] = 0.0f;
    #pragma unroll
    for (int j = 0; j < 6; ++j) {
        const float* vp = qkv + ((size_t)(bb * SEQ + keys[j])) * 3 * D_MODEL + 2 * D_MODEL + h * HD;
        float p = sc[j] * inv;
        for (int c = 0; c < HD; ++c) acc[c] += p * vp[c];
    }
    bf16* op = attn + ((size_t)(bb * SEQ + s)) * D_MODEL + h * HD;
    for (int c = 0; c < HD; ++c) op[c] = (bf16)acc[c];
}

// =====================================================================
// Attention, token rows (s in 0..4): full softmax over 1029 keys.
// One block per (b, h, s): 8*12*5 = 480 blocks.
// =====================================================================
__global__ __launch_bounds__(256) void attn_tok_kernel(
    const float* __restrict__ qkv, bf16* __restrict__ attn)
{
    __shared__ float sc[SEQ];
    __shared__ float red[256];
    __shared__ float qsh[HD];
    const int blk = blockIdx.x;
    const int bb = blk / (NH * 5);
    const int rem = blk % (NH * 5);
    const int h = rem / 5, s = rem % 5;
    const int tid = threadIdx.x;

    if (tid < HD) qsh[tid] = qkv[((size_t)(bb * SEQ + s)) * 3 * D_MODEL + h * HD + tid];
    __syncthreads();

    for (int t = tid; t < SEQ; t += 256) {
        const float* kp = qkv + ((size_t)(bb * SEQ + t)) * 3 * D_MODEL + D_MODEL + h * HD;
        float d = 0.0f;
        for (int c = 0; c < HD; ++c) d += qsh[c] * kp[c];
        sc[t] = d * 0.125f;
    }
    __syncthreads();

    float mx = -1e30f;
    for (int t = tid; t < SEQ; t += 256) mx = fmaxf(mx, sc[t]);
    red[tid] = mx; __syncthreads();
    for (int st = 128; st > 0; st >>= 1) { if (tid < st) red[tid] = fmaxf(red[tid], red[tid + st]); __syncthreads(); }
    mx = red[0]; __syncthreads();

    float ps = 0.0f;
    for (int t = tid; t < SEQ; t += 256) { float e = expf(sc[t] - mx); sc[t] = e; ps += e; }
    red[tid] = ps; __syncthreads();
    for (int st = 128; st > 0; st >>= 1) { if (tid < st) red[tid] += red[tid + st]; __syncthreads(); }
    float inv = 1.0f / red[0]; __syncthreads();

    // probs @ V: 4 key-chunks x 64 dims
    const int d = tid & 63;
    const int chunk = tid >> 6;
    int t0 = chunk * 258; int t1 = t0 + 258; if (t1 > SEQ) t1 = SEQ;
    float part = 0.0f;
    for (int t = t0; t < t1; ++t) {
        const float* vp = qkv + ((size_t)(bb * SEQ + t)) * 3 * D_MODEL + 2 * D_MODEL + h * HD;
        part += sc[t] * vp[d];
    }
    red[tid] = part; __syncthreads();
    if (tid < 64) {
        float o = (red[tid] + red[tid + 64] + red[tid + 128] + red[tid + 192]) * inv;
        attn[((size_t)(bb * SEQ + s)) * D_MODEL + h * HD + tid] = (bf16)o;
    }
}

// ---------------- fp32 -> bf16 weight conversion -----------------------
__global__ void cvt_bf16_kernel(const float* __restrict__ in, bf16* __restrict__ out, int n) {
    int i = blockIdx.x * 256 + threadIdx.x;
    if (i < n) out[i] = (bf16)in[i];
}

// ---------------- final scatter: tokens -> (feat NCHW, ctx, reg) -------
__global__ void scatter_out_kernel(const float* __restrict__ outb, float* __restrict__ dout) {
    const int FEAT = BATCH * D_MODEL * HWSZ;
    const int CTXN = BATCH * D_MODEL;
    const int REGN = BATCH * NREG * D_MODEL;
    int t = blockIdx.x * 256 + threadIdx.x;
    if (t < FEAT) {
        int hw = t & (HWSZ - 1);
        int c  = (t >> 10) % D_MODEL;
        int bb = t / (D_MODEL * HWSZ);
        dout[t] = outb[((size_t)(bb * SEQ + 5 + hw)) * D_MODEL + c];
    } else if (t < FEAT + CTXN) {
        int t2 = t - FEAT; int bb = t2 / D_MODEL, c = t2 % D_MODEL;
        dout[t] = outb[((size_t)(bb * SEQ)) * D_MODEL + c];
    } else if (t < FEAT + CTXN + REGN) {
        int t3 = t - FEAT - CTXN;
        int bb = t3 / (NREG * D_MODEL);
        int r  = (t3 / D_MODEL) % NREG;
        int c  = t3 % D_MODEL;
        dout[t] = outb[((size_t)(bb * SEQ + 1 + r)) * D_MODEL + c];
    }
}

// =====================================================================
extern "C" void kernel_launch(void* const* d_in, const int* in_sizes, int n_in,
                              void* d_out, int out_size, void* d_ws, size_t ws_size,
                              hipStream_t stream)
{
    const float* x    = (const float*)d_in[0];
    const float* ctx  = (const float*)d_in[1];
    const float* regs = (const float*)d_in[2];
    const float* wqkv = (const float*)d_in[3];
    const float* bqkv = (const float*)d_in[4];
    const float* wout = (const float*)d_in[5];
    const float* bout = (const float*)d_in[6];
    const float* ln1g = (const float*)d_in[7];
    const float* ln1b = (const float*)d_in[8];
    const float* ln2g = (const float*)d_in[9];
    const float* ln2b = (const float*)d_in[10];
    const float* w1   = (const float*)d_in[11];
    const float* b1   = (const float*)d_in[12];
    const float* w2   = (const float*)d_in[13];
    const float* b2   = (const float*)d_in[14];

    // ---- workspace carve-up (~255 MB total) ----
    char* ws = (char*)d_ws;
    size_t off = 0;
    auto alloc = [&](size_t bytes) -> void* {
        void* p = ws + off;
        off += (bytes + 255) & ~(size_t)255;
        return p;
    };
    const size_t M = MTOT;
    float* xw    = (float*)alloc(M * D_MODEL * 4);
    bf16*  xwh   = (bf16*) alloc(M * D_MODEL * 2);
    float* qkv   = (float*)alloc(M * 3 * D_MODEL * 4);
    bf16*  attn  = (bf16*) alloc(M * D_MODEL * 2);
    float* h1    = (float*)alloc(M * D_MODEL * 4);
    bf16*  ln2o  = (bf16*) alloc(M * D_MODEL * 2);
    bf16*  ff1   = (bf16*) alloc(M * 4 * D_MODEL * 2);
    float* outb  = (float*)alloc(M * D_MODEL * 4);
    bf16*  wqkvh = (bf16*) alloc((size_t)3 * D_MODEL * D_MODEL * 2);
    bf16*  wouth = (bf16*) alloc((size_t)D_MODEL * D_MODEL * 2);
    bf16*  w1h   = (bf16*) alloc((size_t)4 * D_MODEL * D_MODEL * 2);
    bf16*  w2h   = (bf16*) alloc((size_t)4 * D_MODEL * D_MODEL * 2);
    (void)alloc(256);   // guard pad: GEMM staging may over-read <=64B past last buffer

    // ---- weights -> bf16 ----
    {
        int n;
        n = 3 * D_MODEL * D_MODEL;
        cvt_bf16_kernel<<<(n + 255) / 256, 256, 0, stream>>>(wqkv, wqkvh, n);
        n = D_MODEL * D_MODEL;
        cvt_bf16_kernel<<<(n + 255) / 256, 256, 0, stream>>>(wout, wouth, n);
        n = 4 * D_MODEL * D_MODEL;
        cvt_bf16_kernel<<<(n + 255) / 256, 256, 0, stream>>>(w1, w1h, n);
        cvt_bf16_kernel<<<(n + 255) / 256, 256, 0, stream>>>(w2, w2h, n);
    }

    // ---- concat + LN1 ----
    prep_ln1_kernel<<<MTOT, 256, 0, stream>>>(x, ctx, regs, ln1g, ln1b, xw, xwh);

    // ---- QKV GEMM: (8232 x 768) x (2304 x 768)^T -> fp32 ----
    gemm_bf16_wmma<0><<<dim3(3 * D_MODEL / 128, (MTOT + 127) / 128), 256, 0, stream>>>(
        xwh, wqkvh, bqkv, nullptr, (void*)qkv, MTOT, 3 * D_MODEL, D_MODEL);

    // ---- masked attention ----
    attn_feat_kernel<<<(BATCH * HWSZ * NH + 255) / 256, 256, 0, stream>>>(qkv, attn);
    attn_tok_kernel<<<BATCH * NH * 5, 256, 0, stream>>>(qkv, attn);

    // ---- out projection + residual(xw) -> h1 fp32 ----
    gemm_bf16_wmma<1><<<dim3(D_MODEL / 128, (MTOT + 127) / 128), 256, 0, stream>>>(
        attn, wouth, bout, xw, (void*)h1, MTOT, D_MODEL, D_MODEL);

    // ---- LN2 -> bf16 ----
    ln2_kernel<<<MTOT, 256, 0, stream>>>(h1, ln2g, ln2b, ln2o);

    // ---- FFN1 + exact GELU -> bf16 ----
    gemm_bf16_wmma<2><<<dim3(4 * D_MODEL / 128, (MTOT + 127) / 128), 256, 0, stream>>>(
        ln2o, w1h, b1, nullptr, (void*)ff1, MTOT, 4 * D_MODEL, D_MODEL);

    // ---- FFN2 + residual(h1) -> out fp32 ----
    gemm_bf16_wmma<1><<<dim3(D_MODEL / 128, (MTOT + 127) / 128), 256, 0, stream>>>(
        ff1, w2h, b2, h1, (void*)outb, MTOT, D_MODEL, 4 * D_MODEL);

    // ---- scatter to (feat NCHW | ctx | reg) ----
    {
        int total = BATCH * D_MODEL * HWSZ + BATCH * D_MODEL + BATCH * NREG * D_MODEL;
        scatter_out_kernel<<<(total + 255) / 256, 256, 0, stream>>>(outb, (float*)d_out);
    }
}
